// TemEnc_5514738008907
// MI455X (gfx1250) — compile-verified, matrix operand order checked by reading, not run on hardware
//
#include <hip/hip_runtime.h>
#include <hip/hip_bf16.h>

typedef _Float16 half_t;
typedef __attribute__((ext_vector_type(16))) _Float16 v16h;
typedef __attribute__((ext_vector_type(8)))  float    v8f;
typedef __attribute__((ext_vector_type(4)))  unsigned int v4u;
typedef __attribute__((ext_vector_type(8)))  int v8i;
typedef __attribute__((ext_vector_type(4)))  int v4i;

#define BB   8
#define LL   1024
#define CINC 38
#define DM   512
#define ELAY 2
#define SWIN 25
#define TRK  512
#define NH   8
#define DKH  64

// ---------------------------------------------------------------- utilities
__device__ inline float pe_val(int l, int d) {
  int i2 = (d >> 1) * 2;
  float div = expf(-(float)i2 * (logf(10000.0f) / (float)DM));
  float ang = (float)l * div;
  return (d & 1) ? cosf(ang) : sinf(ang);
}

__device__ inline v8f vzero() {
  v8f z = {0.f, 0.f, 0.f, 0.f, 0.f, 0.f, 0.f, 0.f};
  return z;
}

// ---------------------------------------------------------------- WMMA core
__device__ inline v8f wmma32(v16h a, v16h b, v8f c) {
  // v_wmma_f32_16x16x32_f16  D = A*B + C
  return __builtin_amdgcn_wmma_f32_16x16x32_f16(
      /*neg_a=*/false, a, /*neg_b=*/false, b,
      /*c_mod=*/(short)0, c, /*reuse_a=*/false, /*reuse_b=*/false);
}

// K offset of reg-pair v for this lane-half (ISA 16-bit A 16x32 layout)
__device__ inline int koff(int v, int hlf) {
  return ((v >> 2) << 4) + hlf * 8 + (v & 3) * 2;
}

// A-fragment: 16x32 f16 tile of row-major matrix A (lda), rows row0.., cols k0..
// Works for global or LDS-backed pointers (compiler picks global_load/ds_load).
__device__ inline v16h load_frag_rm(const half_t* __restrict__ A, int lda,
                                    int row0, int k0, int r, int hlf) {
  v16h f;
#pragma unroll
  for (int v = 0; v < 8; ++v) {
    int k = koff(v, hlf);
    const half_t* p = A + (size_t)(row0 + r) * lda + (size_t)(k0 + k);
    f[2 * v]     = p[0];
    f[2 * v + 1] = p[1];
  }
  return f;
}

// B-fragment for B stored row-major as [K][N] (ldb = N): B^T loaded with the
// A-fragment pattern (lane r -> column n0+r, reg pairs -> K).
__device__ inline v16h load_fragB_kn(const half_t* __restrict__ Bm, int ldb,
                                     int k0, int n0, int r, int hlf) {
  v16h f;
#pragma unroll
  for (int v = 0; v < 8; ++v) {
    int k = koff(v, hlf);
    const half_t* p = Bm + (size_t)(k0 + k) * ldb + (size_t)(n0 + r);
    f[2 * v]     = p[0];
    f[2 * v + 1] = p[ldb];
  }
  return f;
}

// ---------------------------------------------------------------- TDM
// Issue TENSOR_LOAD_TO_LDS for a 2D f16 tile: `rows` rows x DM halves,
// contiguous row-major source (stride = DM elements). D# packed per ISA
// cdna5 CDNA5 ISA.
__device__ inline void tdm_load_tile(unsigned lds_off, const half_t* src,
                                     int rows) {
  unsigned long long ga = (unsigned long long)(size_t)src;
  v4u g0;
  g0.x = 1u;                                         // count=1, user mode
  g0.y = lds_off;                                    // lds_addr (bytes)
  g0.z = (unsigned)(ga & 0xFFFFFFFFu);               // global_addr[31:0]
  g0.w = (unsigned)((ga >> 32) & 0x01FFFFFFu)        // global_addr[56:32]
         | (2u << 30);                               // type=2 ("image")
  unsigned td0 = (unsigned)DM;                       // tensor_dim0 (elems)
  unsigned td1 = (unsigned)rows;                     // tensor_dim1
  v8i g1;
  g1[0] = (int)(1u << 16);                           // data_size=1 (2 bytes)
  g1[1] = (int)((td0 & 0xFFFFu) << 16);              // tensor_dim0[15:0]
  g1[2] = (int)((td0 >> 16) | ((td1 & 0xFFFFu) << 16));  // dim0 hi | dim1 lo
  g1[3] = (int)((td1 >> 16) | ((unsigned)DM << 16)); // dim1 hi | tile_dim0
  g1[4] = (int)((unsigned)rows & 0xFFFFu);           // tile_dim1 | tile_dim2=0
  g1[5] = (int)DM;                                   // tensor_dim0_stride lo
  g1[6] = 0;                                         // stride0 hi | stride1 lo
  g1[7] = 0;                                         // stride1 hi
  v4i g2 = {0, 0, 0, 0};
  v4i g3 = {0, 0, 0, 0};
#if __clang_major__ >= 23
  v8i g4 = {0, 0, 0, 0, 0, 0, 0, 0};
  __builtin_amdgcn_tensor_load_to_lds(g0, g1, g2, g3, g4, 0);
#else
  __builtin_amdgcn_tensor_load_to_lds(g0, g1, g2, g3, 0);
#endif
}

// ---------------------------------------------------------------- GEMM
// C[M,DM] = act(A[M,DM]*WT[DM,DM]^T + bias) (+resid). 4 waves/block,
// 64x64 tile. The 64xDM weight strip for this block's columns is DMA'd to
// LDS once by the Tensor Data Mover; B-fragments then come from LDS while
// A streams from global (L2-resident). ACT: 0 none, 1 exact GELU, 2 sigmoid.
template <int ACT>
__global__ void gemm_kernel(int M,
                            const half_t* __restrict__ A,
                            const half_t* __restrict__ WT,
                            const float* __restrict__ bias,
                            const float* __restrict__ resid,
                            float* __restrict__ out32,
                            half_t* __restrict__ out16) {
  __shared__ half_t Bs[64 * DM];  // 64 KB weight strip
  int lane = threadIdx.x & 31;
  int wid  = threadIdx.x >> 5;
  int r    = lane & 15;
  int hlf  = lane >> 4;
  int m0   = blockIdx.y * 64 + wid * 16;   // M launched as multiple of 64
  int n0   = blockIdx.x * 64;

  if (wid == 0) {
    // LDS aperture: flat addr[31:0] == LDS byte offset (ISA 10.2)
    unsigned lds_off = (unsigned)(size_t)(&Bs[0]);
    tdm_load_tile(lds_off, WT + (size_t)n0 * DM, 64);
    __builtin_amdgcn_s_wait_tensorcnt(0);
  }
  __syncthreads();

  v8f z = vzero();
  v8f acc[4] = {z, z, z, z};

  for (int k0 = 0; k0 < DM; k0 += 32) {
    // prefetch next A K-tile (global_prefetch_b8)
    __builtin_prefetch(A + (size_t)(m0 + r) * DM + (size_t)(k0 + 32), 0, 1);
    v16h af = load_frag_rm(A, DM, m0, k0, r, hlf);
#pragma unroll
    for (int t = 0; t < 4; ++t) {
      v16h bf = load_frag_rm(Bs, DM, t * 16, k0, r, hlf);  // ds_load path
      acc[t] = wmma32(af, bf, acc[t]);
    }
  }

#pragma unroll
  for (int t = 0; t < 4; ++t) {
    int col = n0 + t * 16 + r;
    float bv = bias ? bias[col] : 0.0f;
#pragma unroll
    for (int i = 0; i < 8; ++i) {
      int row = m0 + hlf * 8 + i;
      float v = acc[t][i] + bv;
      if (ACT == 1) v = 0.5f * v * (1.0f + erff(v * 0.70710678118654752f));
      else if (ACT == 2) v = 1.0f / (1.0f + expf(-v));
      if (resid) v += resid[(size_t)row * DM + col];
      size_t o = (size_t)row * DM + col;
      if (out32) out32[o] = v;
      if (out16) out16[o] = (half_t)v;
    }
  }
}

// ---------------------------------------------------------------- attention
// One wave handles 16 query rows of one (b,h). Scores staged in dynamic LDS.
__global__ void attn_kernel(const half_t* __restrict__ q,
                            const half_t* __restrict__ k,
                            const half_t* __restrict__ v,
                            half_t* __restrict__ o, int Ls) {
  extern __shared__ float Ssm[];  // [16][Ls]
  int lane = threadIdx.x & 31;
  int r    = lane & 15;
  int hlf  = lane >> 4;
  int m0   = blockIdx.x * 16;
  int h    = blockIdx.y;
  int b    = blockIdx.z;

  const half_t* qb = q + ((size_t)b * Ls) * DM + h * DKH;
  const half_t* kb = k + ((size_t)b * Ls) * DM + h * DKH;
  const half_t* vb = v + ((size_t)b * Ls) * DM + h * DKH;

  // phase 1: S = (Q K^T) / sqrt(dk)   (K-dim = 64, two WMMA per 16x16 tile)
  for (int n0 = 0; n0 < Ls; n0 += 16) {
    v8f acc = vzero();
#pragma unroll
    for (int k0 = 0; k0 < DKH; k0 += 32) {
      v16h af = load_frag_rm(qb, DM, m0, k0, r, hlf);
      v16h bf = load_frag_rm(kb, DM, n0, k0, r, hlf);  // rows of K == B^T frag
      acc = wmma32(af, bf, acc);
    }
#pragma unroll
    for (int i = 0; i < 8; ++i)
      Ssm[(size_t)(hlf * 8 + i) * Ls + n0 + r] = acc[i] * 0.125f;
  }
  __syncthreads();

  // phase 2: row softmax (lanes 0..15 own one row each)
  if (lane < 16) {
    float* row = &Ssm[(size_t)lane * Ls];
    float mx = -1e30f;
    for (int j = 0; j < Ls; ++j) mx = fmaxf(mx, row[j]);
    float sum = 0.f;
    for (int j = 0; j < Ls; ++j) { float e = expf(row[j] - mx); row[j] = e; sum += e; }
    float inv = 1.0f / sum;
    for (int j = 0; j < Ls; ++j) row[j] *= inv;
  }
  __syncthreads();

  // phase 3: O = P * V   (P from LDS -> f16 A-frag, V row-major [key][dim])
  v8f z = vzero();
  v8f acc[4] = {z, z, z, z};
  for (int c0 = 0; c0 < Ls; c0 += 32) {
    v16h pf;
#pragma unroll
    for (int vv = 0; vv < 8; ++vv) {
      int kk = koff(vv, hlf);
      pf[2 * vv]     = (half_t)Ssm[(size_t)r * Ls + c0 + kk];
      pf[2 * vv + 1] = (half_t)Ssm[(size_t)r * Ls + c0 + kk + 1];
    }
#pragma unroll
    for (int t = 0; t < 4; ++t) {
      v16h bf = load_fragB_kn(vb, DM, c0, t * 16, r, hlf);
      acc[t] = wmma32(pf, bf, acc[t]);
    }
  }
  half_t* ob = o + ((size_t)b * Ls) * DM + h * DKH;
#pragma unroll
  for (int t = 0; t < 4; ++t)
#pragma unroll
    for (int i = 0; i < 8; ++i)
      ob[(size_t)(m0 + hlf * 8 + i) * DM + t * 16 + r] = (half_t)acc[t][i];
}

// ---------------------------------------------------------------- embedding
__global__ void embed_kernel(const float* __restrict__ x,
                             const float* __restrict__ w,
                             float* __restrict__ ex32) {
  int b = blockIdx.x / LL, l = blockIdx.x % LL;
  __shared__ float xs[3 * CINC];
  int tid = threadIdx.x;
  if (tid < 3 * CINC) {
    int tt = tid / CINC, c = tid % CINC;
    int pos = (l - 1 + tt + LL) % LL;  // circular pad
    xs[tt * CINC + c] = x[((size_t)b * LL + pos) * CINC + c];
  }
  __syncthreads();
  for (int d = tid; d < DM; d += blockDim.x) {
    float acc = 0.f;
    const float* wd = w + (size_t)d * CINC * 3;
#pragma unroll 2
    for (int c = 0; c < CINC; ++c)
      for (int t = 0; t < 3; ++t)
        acc += wd[c * 3 + t] * xs[t * CINC + c];
    acc += pe_val(l, d);
    ex32[((size_t)b * LL + l) * DM + d] = acc;
  }
}

// ---------------------------------------------------------------- score
__global__ void score_kernel(const float* __restrict__ ex,
                             float* __restrict__ score) {
  int b = blockIdx.x / LL, l = blockIdx.x % LL;
  int tid = threadIdx.x;
  int cnt = (l + 1 < SWIN) ? (l + 1) : SWIN;
  float inv = 1.0f / (float)cnt;
  float pv = 0.f, pm = 0.f;
  for (int d = tid; d < DM; d += blockDim.x) {
    float s1 = 0.f, s2 = 0.f;
    for (int w = 0; w < cnt; ++w) {
      float vv = ex[((size_t)b * LL + (l - w)) * DM + d];
      s1 += vv; s2 += vv * vv;
    }
    float m1 = s1 * inv;
    pv += s2 * inv - m1 * m1;
    pm += m1;
  }
  __shared__ float rv[256], rm[256];
  rv[tid] = pv; rm[tid] = pm;
  __syncthreads();
  for (int s = 128; s > 0; s >>= 1) {
    if (tid < s) { rv[tid] += rv[tid + s]; rm[tid] += rm[tid + s]; }
    __syncthreads();
  }
  if (tid == 0) score[blockIdx.x] = rv[0] / rm[0];
}

// ---------------------------------------------------------------- top-k
__global__ void topk_kernel(const float* __restrict__ score,
                            int* __restrict__ midx, int* __restrict__ uidx) {
  __shared__ unsigned long long keys[LL];
  int b = blockIdx.x, tid = threadIdx.x;
  unsigned int u = __float_as_uint(score[(size_t)b * LL + tid]);
  u = (u & 0x80000000u) ? ~u : (u | 0x80000000u);  // order-preserving map
  keys[tid] = ((unsigned long long)u << 32) | (unsigned int)(LL - 1 - tid);
  for (int k = 2; k <= LL; k <<= 1)
    for (int j = k >> 1; j > 0; j >>= 1) {
      __syncthreads();
      int ixj = tid ^ j;
      if (ixj > tid) {
        unsigned long long a = keys[tid], c = keys[ixj];
        if ((a < c) == ((tid & k) == 0)) { keys[tid] = c; keys[ixj] = a; }
      }
    }
  __syncthreads();
  if (tid < TRK)
    midx[(size_t)b * TRK + tid] = LL - 1 - (int)(keys[tid] & 0xFFFFFFFFu);
  if (tid < LL - TRK)
    uidx[(size_t)b * (LL - TRK) + tid] =
        LL - 1 - (int)(keys[LL - 1 - tid] & 0xFFFFFFFFu);
}

// ---------------------------------------------------------------- gather / scatter
__global__ void gather_kernel(const float* __restrict__ ex,
                              const int* __restrict__ uidx,
                              float* __restrict__ um32,
                              half_t* __restrict__ um16) {
  int nu = LL - TRK;
  int b = blockIdx.x / nu, t = blockIdx.x % nu;
  int src = uidx[(size_t)b * nu + t];
  for (int d = threadIdx.x; d < DM; d += blockDim.x) {
    float vv = ex[((size_t)b * LL + src) * DM + d];
    size_t o = ((size_t)b * nu + t) * DM + d;
    um32[o] = vv; um16[o] = (half_t)vv;
  }
}

__global__ void scatter_kernel(const float* __restrict__ ux,
                               const int* __restrict__ uidx,
                               const int* __restrict__ midx,
                               const float* __restrict__ mask_tok,
                               float* __restrict__ tok32,
                               half_t* __restrict__ tok16) {
  int nu = LL - TRK;
  int b = blockIdx.x / LL, slot = blockIdx.x % LL;
  if (slot < nu) {
    int dst = uidx[(size_t)b * nu + slot];
    for (int d = threadIdx.x; d < DM; d += blockDim.x) {
      float vv = ux[((size_t)b * nu + slot) * DM + d];
      size_t o = ((size_t)b * LL + dst) * DM + d;
      tok32[o] = vv; tok16[o] = (half_t)vv;
    }
  } else {
    int dst = midx[(size_t)b * TRK + (slot - nu)];
    for (int d = threadIdx.x; d < DM; d += blockDim.x) {
      float vv = mask_tok[d] + pe_val(dst, d);
      size_t o = ((size_t)b * LL + dst) * DM + d;
      tok32[o] = vv; tok16[o] = (half_t)vv;
    }
  }
}

// ---------------------------------------------------------------- layernorm
__global__ void ln_kernel(const float* __restrict__ xin,
                          const float* __restrict__ g,
                          const float* __restrict__ bb,
                          float* __restrict__ o32, half_t* __restrict__ o16) {
  int row = blockIdx.x, tid = threadIdx.x;
  const float* xr = xin + (size_t)row * DM;
  float s = 0.f, s2 = 0.f;
  for (int d = tid; d < DM; d += blockDim.x) {
    float vv = xr[d]; s += vv; s2 += vv * vv;
  }
  __shared__ float rs[256], rq[256];
  rs[tid] = s; rq[tid] = s2;
  __syncthreads();
  for (int st = 128; st > 0; st >>= 1) {
    if (tid < st) { rs[tid] += rs[tid + st]; rq[tid] += rq[tid + st]; }
    __syncthreads();
  }
  float mean = rs[0] / (float)DM;
  float var  = rq[0] / (float)DM - mean * mean;
  float inv  = rsqrtf(var + 1e-5f);
  for (int d = tid; d < DM; d += blockDim.x) {
    float vv = (xr[d] - mean) * inv * g[d] + bb[d];
    size_t o = (size_t)row * DM + d;
    if (o32) o32[o] = vv;
    if (o16) o16[o] = (half_t)vv;
  }
}

// ---------------------------------------------------------------- weight prep
// in f32 [nmat,Din,Dout] -> out f16 [nmat,Dout,Din]  (transpose + convert)
__global__ void wprep_kernel(const float* __restrict__ win,
                             half_t* __restrict__ wout, int nmat) {
  int idx = blockIdx.x * blockDim.x + threadIdx.x;
  int tot = nmat * DM * DM;
  if (idx >= tot) return;
  int m = idx / (DM * DM);
  int rr = idx % (DM * DM);
  int n = rr / DM, k = rr % DM;
  wout[idx] = (half_t)win[(size_t)m * DM * DM + (size_t)k * DM + n];
}

// ---------------------------------------------------------------- host side
static inline void launch_gemm(hipStream_t st, int M,
                               const half_t* A, const half_t* WT,
                               const float* bias, const float* resid,
                               float* o32, half_t* o16, int act) {
  dim3 g(DM / 64, M / 64), blk(128);
  if (act == 0)      gemm_kernel<0><<<g, blk, 0, st>>>(M, A, WT, bias, resid, o32, o16);
  else if (act == 1) gemm_kernel<1><<<g, blk, 0, st>>>(M, A, WT, bias, resid, o32, o16);
  else               gemm_kernel<2><<<g, blk, 0, st>>>(M, A, WT, bias, resid, o32, o16);
}

static void run_stack(hipStream_t st, float* x32, half_t* x16, int Bn, int Ls,
                      const half_t* wt, const float* bias,
                      const float* g, const float* bb,
                      half_t* q, half_t* k, half_t* v, half_t* attn) {
  int M = Bn * Ls;
  for (int l = 0; l < ELAY; ++l) {
    const half_t* W = wt + (size_t)l * 4 * DM * DM;
    const float* bp = bias + (size_t)l * 4 * DM;
    launch_gemm(st, M, x16, W + 0 * (size_t)DM * DM, bp + 0 * DM, nullptr, nullptr, q, 0);
    launch_gemm(st, M, x16, W + 1 * (size_t)DM * DM, bp + 1 * DM, nullptr, nullptr, k, 0);
    launch_gemm(st, M, x16, W + 2 * (size_t)DM * DM, bp + 2 * DM, nullptr, nullptr, v, 0);
    attn_kernel<<<dim3(Ls / 16, NH, Bn), 32, (size_t)16 * Ls * sizeof(float), st>>>(q, k, v, attn, Ls);
    // x = x + attn @ Wo + bo  (residual fused in epilogue)
    launch_gemm(st, M, attn, W + 3 * (size_t)DM * DM, bp + 3 * DM, x32, x32, x16, 0);
  }
  ln_kernel<<<M, 256, 0, st>>>(x32, g, bb, x32, x16);
}

extern "C" void kernel_launch(void* const* d_in, const int* in_sizes, int n_in,
                              void* d_out, int out_size, void* d_ws, size_t ws_size,
                              hipStream_t stream) {
  (void)in_sizes; (void)n_in; (void)out_size; (void)ws_size;
  const float* x        = (const float*)d_in[0];
  const float* emb_w    = (const float*)d_in[1];
  const float* enc_w    = (const float*)d_in[2];
  const float* enc_b    = (const float*)d_in[3];
  const float* enc_g    = (const float*)d_in[4];
  const float* enc_bb   = (const float*)d_in[5];
  const float* dec_w    = (const float*)d_in[6];
  const float* dec_b    = (const float*)d_in[7];
  const float* dec_g    = (const float*)d_in[8];
  const float* dec_bb   = (const float*)d_in[9];
  const float* mask_tok = (const float*)d_in[10];
  const float* pw1      = (const float*)d_in[11];
  const float* pb1      = (const float*)d_in[12];
  const float* pw2      = (const float*)d_in[13];
  const float* pb2      = (const float*)d_in[14];

  char* ws = (char*)d_ws;
  size_t off = 0;
  auto alloc = [&](size_t bytes) -> void* {
    void* p = ws + off;
    off = (off + bytes + 255) & ~(size_t)255;
    return p;
  };

  const size_t NTOK = (size_t)BB * LL;         // 8192
  const size_t NU   = (size_t)BB * (LL - TRK); // 4096 unmasked tokens

  float*  ex32   = (float*) alloc(NTOK * DM * sizeof(float));   // reused as tok32
  float*  score  = (float*) alloc(NTOK * sizeof(float));
  int*    midx   = (int*)   alloc((size_t)BB * TRK * sizeof(int));
  int*    uidx   = (int*)   alloc((size_t)BB * (LL - TRK) * sizeof(int));
  float*  um32   = (float*) alloc(NU * DM * sizeof(float));
  half_t* um16   = (half_t*)alloc(NU * DM * sizeof(half_t));
  half_t* tok16  = (half_t*)alloc(NTOK * DM * sizeof(half_t));
  half_t* q16    = (half_t*)alloc(NTOK * DM * sizeof(half_t));  // reused as h1
  half_t* k16    = (half_t*)alloc(NTOK * DM * sizeof(half_t));
  half_t* v16    = (half_t*)alloc(NTOK * DM * sizeof(half_t));
  half_t* attn16 = (half_t*)alloc(NTOK * DM * sizeof(half_t));
  half_t* wtenc  = (half_t*)alloc((size_t)ELAY * 4 * DM * DM * sizeof(half_t));
  half_t* wtdec  = (half_t*)alloc((size_t)ELAY * 4 * DM * DM * sizeof(half_t));
  half_t* wtp1   = (half_t*)alloc((size_t)DM * DM * sizeof(half_t));
  half_t* wtp2   = (half_t*)alloc((size_t)DM * DM * sizeof(half_t));
  float*  tok32  = ex32;   // ex dead after gather/score -> reuse for tokens
  half_t* h1_16  = q16;    // q dead after decoder -> reuse for proj hidden

  // 1. weights: f32 [.,K,N] -> f16 [.,N,K]
  int totw = ELAY * 4 * DM * DM;
  wprep_kernel<<<(totw + 255) / 256, 256, 0, stream>>>(enc_w, wtenc, ELAY * 4);
  wprep_kernel<<<(totw + 255) / 256, 256, 0, stream>>>(dec_w, wtdec, ELAY * 4);
  wprep_kernel<<<(DM * DM + 255) / 256, 256, 0, stream>>>(pw1, wtp1, 1);
  wprep_kernel<<<(DM * DM + 255) / 256, 256, 0, stream>>>(pw2, wtp2, 1);

  // 2. embedding (circular conv + PE)
  embed_kernel<<<BB * LL, 256, 0, stream>>>(x, emb_w, ex32);

  // 3. windowed variance score
  score_kernel<<<BB * LL, 256, 0, stream>>>(ex32, score);

  // 4. per-batch bitonic top-k split
  topk_kernel<<<BB, LL, 0, stream>>>(score, midx, uidx);

  // 5. gather unmasked tokens
  gather_kernel<<<(int)NU, 256, 0, stream>>>(ex32, uidx, um32, um16);

  // 6. encoder on unmasked half (L=512)
  run_stack(stream, um32, um16, BB, LL - TRK, wtenc, enc_b, enc_g, enc_bb,
            q16, k16, v16, attn16);

  // 7. scatter encoder output + mask tokens into full sequence
  scatter_kernel<<<BB * LL, 256, 0, stream>>>(um32, uidx, midx, mask_tok,
                                              tok32, tok16);

  // 8. decoder on full sequence (L=1024)
  run_stack(stream, tok32, tok16, BB, LL, wtdec, dec_b, dec_g, dec_bb,
            q16, k16, v16, attn16);

  // 9. projection head: sigmoid(gelu(dx@W1+b1)@W2+b2)
  launch_gemm(stream, BB * LL, tok16, wtp1, pb1, nullptr, nullptr, h1_16, 1);
  launch_gemm(stream, BB * LL, h1_16, wtp2, pb2, nullptr,
              (float*)d_out, nullptr, 2);
}
